// Attention_86182813762217
// MI455X (gfx1250) — matrix operand order repaired and checked by reading, not compiled, not run on hardware
//
#include <hip/hip_runtime.h>
#include <hip/hip_bf16.h>

typedef __attribute__((ext_vector_type(16))) _Float16 v16h;
typedef __attribute__((ext_vector_type(8)))  _Float16 v8h;
typedef __attribute__((ext_vector_type(8)))  float    v8f;
typedef __attribute__((ext_vector_type(4)))  float    v4f;

#define B_  4
#define T_  4096
#define C_  1024
#define H_  64

__device__ __forceinline__ v8f wmma_f16(v16h a, v16h b, v8f c) {
    return __builtin_amdgcn_wmma_f32_16x16x32_f16(
        /*neg_a=*/false, a, /*neg_b=*/false, b,
        /*c_mod=*/(short)0, c, /*reuse_a=*/false, /*reuse_b=*/false);
}

// ---------------------------------------------------------------------------
// Kernel 0: convert+transpose weights: Wt[mat][n][k] (f16) from W[k][n] (f32)
// ---------------------------------------------------------------------------
__global__ void wt_convert_kernel(const float* __restrict__ Wq,
                                  const float* __restrict__ Wk,
                                  const float* __restrict__ Wv,
                                  _Float16* __restrict__ Wt) {
    int idx = blockIdx.x * 256 + threadIdx.x;   // 3*64*1024 = 196608 total
    if (idx >= 3 * H_ * C_) return;
    int k   = idx & (C_ - 1);
    int n   = (idx >> 10) & (H_ - 1);
    int mat = idx >> 16;
    const float* W = (mat == 0) ? Wq : (mat == 1) ? Wk : Wv;
    Wt[idx] = (_Float16)W[(size_t)k * H_ + n];
}

// ---------------------------------------------------------------------------
// Kernel 1: projections. One wave computes a 16-row x 64-col tile of Q,K,V.
// Qh,Kh stored row-major [B*T][64] f16; V stored transposed Vt[B][64][T] f16.
// B fragments for one matrix are batched (4x v16h) ahead of the 4 WMMAs so
// the scheduler can stage s_wait_loadcnt instead of waiting to zero each time.
// ---------------------------------------------------------------------------
__global__ void __launch_bounds__(128)
proj_kernel(const float* __restrict__ X, const _Float16* __restrict__ Wt,
            _Float16* __restrict__ Qh, _Float16* __restrict__ Kh,
            _Float16* __restrict__ Vt) {
    const int lane = threadIdx.x & 31;
    const int wave = threadIdx.x >> 5;
    const int tile = blockIdx.x * 4 + wave;     // 0..1023
    const int row0 = tile * 16;                 // global row in [0, B*T)
    const int m    = lane & 15;
    const int hh   = lane >> 4;

    v8f acc[3][4];
    const v8f vzero = {0.f,0.f,0.f,0.f,0.f,0.f,0.f,0.f};
#pragma unroll
    for (int a = 0; a < 3; ++a)
#pragma unroll
        for (int c = 0; c < 4; ++c) acc[a][c] = vzero;

    const float* xrow = X + (size_t)(row0 + m) * C_;

    for (int kk = 0; kk < C_; kk += 32) {
        // A fragment of X (16x32, f16): lane m holds K run [hh*16..+15]
        const int o1 = kk + hh * 16;
        const int o2 = kk + hh * 16 + 8;
        v4f x0 = *(const v4f*)(xrow + o1);
        v4f x1 = *(const v4f*)(xrow + o1 + 4);
        v4f x2 = *(const v4f*)(xrow + o2);
        v4f x3 = *(const v4f*)(xrow + o2 + 4);
        v16h a;
#pragma unroll
        for (int j = 0; j < 4; ++j) {
            a[j]      = (_Float16)x0[j];
            a[4 + j]  = (_Float16)x1[j];
            a[8 + j]  = (_Float16)x2[j];
            a[12 + j] = (_Float16)x3[j];
        }
#pragma unroll
        for (int mat = 0; mat < 3; ++mat) {
            // Batch all 4 B fragments (32x16 each) for this matrix first.
            v16h bf[4];
#pragma unroll
            for (int c = 0; c < 4; ++c) {
                const _Float16* wp =
                    Wt + ((size_t)(mat * H_ + c * 16 + m)) * C_ + kk + hh * 16;
                bf[c] = *(const v16h*)wp;
            }
#pragma unroll
            for (int c = 0; c < 4; ++c)
                acc[mat][c] = wmma_f16(a, bf[c], acc[mat][c]);
        }
    }

    // Store: C/D layout -> lane holds col n = m, rows r + 8*hh
    const int b     = row0 >> 12;       // T_ = 4096
    const int tloc0 = row0 & (T_ - 1);
#pragma unroll
    for (int c = 0; c < 4; ++c) {
#pragma unroll
        for (int r = 0; r < 8; ++r) {
            const int Mrow = r + 8 * hh;
            const size_t grow = (size_t)(row0 + Mrow);
            Qh[grow * H_ + c * 16 + m] = (_Float16)acc[0][c][r];
            Kh[grow * H_ + c * 16 + m] = (_Float16)acc[1][c][r];
            Vt[((size_t)(b * H_ + c * 16 + m)) * T_ + (tloc0 + Mrow)] =
                (_Float16)acc[2][c][r];
        }
    }
}

// ---------------------------------------------------------------------------
// Kernel 2: fused causal flash attention. One wave per 16-query tile.
// ---------------------------------------------------------------------------
__global__ void __launch_bounds__(128)
attn_kernel(const _Float16* __restrict__ Qh, const _Float16* __restrict__ Kh,
            const _Float16* __restrict__ Vt, float* __restrict__ out) {
    __shared__ _Float16 smem[4][16 * 32];   // per-wave P-transpose staging
    const int lane = threadIdx.x & 31;
    const int wave = threadIdx.x >> 5;
    const int tile = blockIdx.x * 4 + wave;       // 0..1023
    const int b    = tile >> 8;                   // 256 query tiles per batch
    const int q0   = (tile & 255) * 16;
    const int m    = lane & 15;
    const int hh   = lane >> 4;

    // Q A-fragments for head-dim K blocks [0..31] and [32..63]
    const _Float16* qrow = Qh + ((size_t)(b * T_ + q0 + m)) * H_;
    v16h Aq[2];
#pragma unroll
    for (int kb = 0; kb < 2; ++kb) {
        v8h lo = *(const v8h*)(qrow + kb * 32 + hh * 16);
        v8h hi = *(const v8h*)(qrow + kb * 32 + hh * 16 + 8);
#pragma unroll
        for (int j = 0; j < 8; ++j) { Aq[kb][j] = lo[j]; Aq[kb][8 + j] = hi[j]; }
    }

    const v8f vzero = {0.f,0.f,0.f,0.f,0.f,0.f,0.f,0.f};
    v8f O[4];
#pragma unroll
    for (int c = 0; c < 4; ++c) O[c] = vzero;
    float mrow[8], lrow[8];
#pragma unroll
    for (int r = 0; r < 8; ++r) { mrow[r] = -__builtin_inff(); lrow[r] = 0.f; }

    const float scale = 0.125f;            // H^-0.5
    const int nk = q0 + 16;                // keys needed (causal)
    const _Float16* Kb = Kh + (size_t)b * T_ * H_;
    const _Float16* Vb = Vt + (size_t)b * H_ * T_;

    for (int s0 = 0; s0 < nk; s0 += 32) {
        // ---- scores: two 16x16 tiles (keys s0..+15 and s0+16..+31) ----
        v8f S0 = vzero, S1 = vzero;
        const _Float16* k0 = Kb + ((size_t)(s0 + m)) * H_ + hh * 16;
        if (s0 + 32 < nk) __builtin_prefetch(k0 + 32 * H_, 0, 1);
        v16h B0a = *(const v16h*)(k0);
        v16h B0b = *(const v16h*)(k0 + 32);
        const _Float16* k1 = k0 + (size_t)16 * H_;
        v16h B1a = *(const v16h*)(k1);
        v16h B1b = *(const v16h*)(k1 + 32);
        S0 = wmma_f16(Aq[0], B0a, S0);
        S0 = wmma_f16(Aq[1], B0b, S0);
        S1 = wmma_f16(Aq[0], B1a, S1);
        S1 = wmma_f16(Aq[1], B1b, S1);

        // ---- scale + causal mask + online softmax (rows across 16-lane halves)
        const bool need_mask = (s0 + 31 > q0);   // wave-uniform
        float p0[8], p1[8], alpha[8];
#pragma unroll
        for (int r = 0; r < 8; ++r) {
            const int row = q0 + r + 8 * hh;
            float a = S0[r] * scale;
            float c = S1[r] * scale;
            if (need_mask) {
                if (s0 + m > row)      a = -__builtin_inff();
                if (s0 + 16 + m > row) c = -__builtin_inff();
            }
            float v = fmaxf(a, c);
#pragma unroll
            for (int off = 1; off < 16; off <<= 1)
                v = fmaxf(v, __shfl_xor(v, off, 32));
            const float mnew = fmaxf(mrow[r], v);
            alpha[r] = __expf(mrow[r] - mnew);
            mrow[r]  = mnew;
            float e0 = __expf(a - mnew);
            float e1 = __expf(c - mnew);
            p0[r] = e0; p1[r] = e1;
            float sum = e0 + e1;
#pragma unroll
            for (int off = 1; off < 16; off <<= 1)
                sum += __shfl_xor(sum, off, 32);
            lrow[r] = lrow[r] * alpha[r] + sum;
        }
#pragma unroll
        for (int c = 0; c < 4; ++c)
#pragma unroll
            for (int r = 0; r < 8; ++r) O[c][r] *= alpha[r];

        // ---- transpose P (C/D layout -> A layout) via wave-private LDS ----
        _Float16* L = smem[wave];
#pragma unroll
        for (int r = 0; r < 8; ++r) {
            const int Mrow = r + 8 * hh;
            L[Mrow * 32 + m]      = (_Float16)p0[r];
            L[Mrow * 32 + 16 + m] = (_Float16)p1[r];
        }
        asm volatile("s_wait_dscnt 0x0" ::: "memory");
        v16h Ap;
        {
            const _Float16* lr = L + m * 32;
            v8h lo = *(const v8h*)(lr + hh * 16);
            v8h hi = *(const v8h*)(lr + hh * 16 + 8);
#pragma unroll
            for (int j = 0; j < 8; ++j) { Ap[j] = lo[j]; Ap[8 + j] = hi[j]; }
        }

        // ---- P (16x32) x V (32x64): B fragments contiguous over keys in Vt
        const _Float16* vb = Vb + ((size_t)m) * T_ + s0 + hh * 16;
        v16h Bv[4];
#pragma unroll
        for (int c = 0; c < 4; ++c)
            Bv[c] = *(const v16h*)(vb + (size_t)(c * 16) * T_);
#pragma unroll
        for (int c = 0; c < 4; ++c)
            O[c] = wmma_f16(Ap, Bv[c], O[c]);
    }

    // ---- normalize and store fp32 output [B,T,64] ----
#pragma unroll
    for (int r = 0; r < 8; ++r) {
        const float inv = 1.0f / lrow[r];
        const size_t grow = (size_t)(b * T_ + q0 + r + 8 * hh);
#pragma unroll
        for (int c = 0; c < 4; ++c)
            out[grow * H_ + c * 16 + m] = O[c][r] * inv;
    }
}

// ---------------------------------------------------------------------------
extern "C" void kernel_launch(void* const* d_in, const int* in_sizes, int n_in,
                              void* d_out, int out_size, void* d_ws, size_t ws_size,
                              hipStream_t stream) {
    const float* X  = (const float*)d_in[0];
    const float* Wq = (const float*)d_in[1];
    const float* Wk = (const float*)d_in[2];
    const float* Wv = (const float*)d_in[3];
    float* out = (float*)d_out;

    char* ws = (char*)d_ws;
    const size_t sz_Wt  = (size_t)3 * H_ * C_ * sizeof(_Float16);   // 384 KB
    const size_t sz_QKV = (size_t)B_ * T_ * H_ * sizeof(_Float16);  // 2 MB each
    _Float16* Wt = (_Float16*)(ws);
    _Float16* Qh = (_Float16*)(ws + sz_Wt);
    _Float16* Kh = (_Float16*)(ws + sz_Wt + sz_QKV);
    _Float16* Vt = (_Float16*)(ws + sz_Wt + 2 * sz_QKV);

    wt_convert_kernel<<<(3 * H_ * C_ + 255) / 256, 256, 0, stream>>>(Wq, Wk, Wv, Wt);
    proj_kernel<<<(B_ * T_ / 16) / 4, 128, 0, stream>>>(X, Wt, Qh, Kh, Vt);
    attn_kernel<<<(B_ * (T_ / 16)) / 4, 128, 0, stream>>>(Qh, Kh, Vt, out);
}
